// RecurrentModule_86311662780493
// MI455X (gfx1250) — matrix operand order repaired and checked by reading, not compiled, not run on hardware
//
#include <hip/hip_runtime.h>
#include <hip/hip_bf16.h>

typedef __attribute__((ext_vector_type(16))) __bf16 v16bf;
typedef __attribute__((ext_vector_type(8)))  float  v8f;

#define N_   32
#define C_   256
#define L_   4096
#define H_   256
#define G_   (4*H_)        // 1024 gate columns
#define K_   (C_+H_)       // 512 combined K
#define NBLK 16            // persistent blocks (1 per WGP), each owns 16 h-columns
#define ZW   (K_+8)        // padded LDS row stride (bf16 elems): 1040B = 65*16, keeps 16B align

// workspace layout (bytes)
#define XTB_BYTES  ((size_t)L_*N_*C_*2)            // 64 MB bf16 x transposed (L,N,C)
#define WB_BYTES   ((size_t)G_*K_*2)               // 1 MB bf16 [W_ih|W_hh]
#define BIAS_BYTES ((size_t)G_*4)
#define HBUF_BYTES ((size_t)2*N_*H_*2)             // double-buffered bf16 h
#define BAR_OFF    (XTB_BYTES + WB_BYTES + BIAS_BYTES + HBUF_BYTES)

__device__ __forceinline__ unsigned short f2bf(float f) {
  union { float f; unsigned u; } v; v.f = f;
  unsigned r = v.u + 0x7FFFu + ((v.u >> 16) & 1u);   // round-to-nearest-even
  return (unsigned short)(r >> 16);
}

__device__ __forceinline__ float fast_sigmoid(float x) {
  return 1.0f / (1.0f + __expf(-x));
}
__device__ __forceinline__ float fast_tanh(float x) {
#if __has_builtin(__builtin_amdgcn_tanhf)
  return __builtin_amdgcn_tanhf(x);        // v_tanh_f32 (TRANS)
#else
  float t = __expf(-2.0f * x);
  return (1.0f - t) / (1.0f + t);
#endif
}

// CDNA5 async memory -> LDS copy (16B per lane), tracked by ASYNCcnt
__device__ __forceinline__ void async_copy_b128(unsigned lds_off, unsigned gvoff, const void* sbase) {
  asm volatile("global_load_async_to_lds_b128 %0, %1, %2"
               :: "v"(lds_off), "v"(gvoff), "s"(sbase) : "memory");
}
__device__ __forceinline__ void wait_async0() {
  asm volatile("s_wait_asynccnt 0" ::: "memory");
}

// ---------------- prep: weights->bf16, bias sum, zero h/barrier ----------------
__global__ void prep_kernel(const float* __restrict__ Wih, const float* __restrict__ Whh,
                            const float* __restrict__ bih, const float* __restrict__ bhh,
                            unsigned short* __restrict__ wb, float* __restrict__ bias,
                            unsigned short* __restrict__ hbuf, unsigned* __restrict__ bar) {
  int idx = blockIdx.x * blockDim.x + threadIdx.x;
  if (idx < G_ * K_) {
    int g = idx >> 9, k = idx & (K_ - 1);
    float w = (k < C_) ? Wih[g * C_ + k] : Whh[g * H_ + (k - C_)];
    wb[idx] = f2bf(w);
  }
  if (idx < G_)        bias[idx] = bih[idx] + bhh[idx];
  if (idx < 2*N_*H_)   hbuf[idx] = 0;
  if (idx < 2)         bar[idx]  = 0;
}

// ---------------- transpose: x (N,C,L) f32 -> xtb (L,N,C) bf16 ----------------
__global__ void transpose_kernel(const float* __restrict__ x, unsigned short* __restrict__ xtb) {
  __shared__ unsigned short tile[64][72];
  const int lb = blockIdx.x, cb = blockIdx.y, n = blockIdx.z;
  const int t = threadIdx.x;
  const int l0 = lb * 64, c0 = cb * 64;
  #pragma unroll
  for (int i = 0; i < 16; ++i) {
    int idx = i * 256 + t;
    int cl = idx >> 6, ll = idx & 63;                   // coalesced along L
    tile[cl][ll] = f2bf(x[((size_t)n * C_ + (c0 + cl)) * L_ + l0 + ll]);
  }
  __syncthreads();
  #pragma unroll
  for (int i = 0; i < 16; ++i) {
    int idx = i * 256 + t;
    int ll = idx >> 6, cl = idx & 63;                   // coalesced along C
    xtb[(size_t)(l0 + ll) * (N_*C_) + (size_t)n * C_ + c0 + cl] = tile[cl][ll];
  }
}

// ---------------- persistent fused LSTM ----------------
// gates(32 x 1024) = [x_t | h] (32 x 512) * Wc^T, block b owns h-cols [16b,16b+16)
// wave w: row-block r=w&1 (batch rows 16r..), gate section s=w>>1 (i,f,g,o)
__global__ void __launch_bounds__(256, 1)
lstm_kernel(const unsigned short* __restrict__ xtb, const unsigned short* __restrict__ wb,
            const float* __restrict__ bias, unsigned short* __restrict__ hbuf,
            unsigned* __restrict__ bar, float* __restrict__ out) {
  __shared__ __align__(16) unsigned short Z[N_][ZW];        // [x_t | h] staged, bf16
  __shared__ __align__(16) float gbuf[4][N_][17];           // gate pre-activations
  __shared__ __align__(16) float cbuf[N_][16];              // cell state
  __shared__ __align__(16) float houts[N_][16][16];         // h staged for 16 steps

  const int tid  = threadIdx.x;
  const int lane = tid & 31;
  const int wave = tid >> 5;
  const int r    = wave & 1;
  const int s    = wave >> 1;
  const int hb0  = blockIdx.x * 16;

  const int half = lane >> 4;       // A-operand lane-half (K sub-block)
  const int mrow = lane & 15;       // A row index
  const int ncol = lane & 15;       // B/D column
  const int kh   = lane >> 4;       // B-operand K half

  // ---- hoist ALL B fragments (this wave's 16 gate columns, full K=512) into VGPRs ----
  union bf_u { v16bf v; unsigned u[8]; };
  bf_u Bf[16];                                 // 128 VGPRs, loop-invariant for 4096 steps
  {
    const unsigned short* wcol = wb + (size_t)(s * H_ + hb0 + ncol) * K_;
    #pragma unroll
    for (int t = 0; t < 16; ++t) {
      #pragma unroll
      for (int q = 0; q < 8; ++q)              // lanes 0-15: K=0..15, lanes 16-31: K=16..31
        Bf[t].u[q] = *(const unsigned*)&wcol[t * 32 + kh * 16 + 2 * q];
    }
  }

  for (int idx = tid; idx < N_ * 16; idx += 256) cbuf[idx >> 4][idx & 15] = 0.0f;
  const float bias_val = bias[s * H_ + hb0 + ncol];
  const unsigned zb = (unsigned)(size_t)&Z[0][0];   // LDS offset = low 32 bits of flat addr
  const unsigned short* zrow = &Z[r * 16 + mrow][0];
  __syncthreads();

  int p = 0;
  for (int l = 0; l < L_; ++l) {
    // ---- async stage Z = [x_t | h_{l-1}] directly into LDS (ASYNCcnt path) ----
    const void* xslice = (const void*)(xtb + (size_t)l * (N_*C_));
    const void* hslice = (const void*)(hbuf + (size_t)p * (N_*H_));
    #pragma unroll
    for (int i = 0; i < 4; ++i) {              // x: 1024 x 16B chunks
      int q = tid + i * 256;
      unsigned lds = zb + (unsigned)(q >> 5) * (ZW * 2) + (unsigned)(q & 31) * 16;
      async_copy_b128(lds, (unsigned)q * 16, xslice);
    }
    #pragma unroll
    for (int i = 0; i < 4; ++i) {              // h: 1024 x 16B chunks -> cols 256..511
      int q = tid + i * 256;
      unsigned lds = zb + (unsigned)(q >> 5) * (ZW * 2) + 512 + (unsigned)(q & 31) * 16;
      async_copy_b128(lds, (unsigned)q * 16, hslice);
    }
    if (l + 1 < L_)  // warm L2 for next timestep's x tile (global_prefetch_b8)
      __builtin_prefetch(xtb + (size_t)(l + 1) * (N_*C_) + tid * 32, 0, 1);
    wait_async0();
    __syncthreads();

    // ---- WMMA: 16x16 tile, K=512 fully unrolled (B frags already in registers) ----
    v8f acc;
    #pragma unroll
    for (int i = 0; i < 8; ++i) acc[i] = bias_val;

    #pragma unroll
    for (int t = 0; t < 16; ++t) {
      const int kc = t * 32;
      union { v16bf v; unsigned u[8]; } A;
      #pragma unroll
      for (int q = 0; q < 4; ++q) {            // two 16B LDS reads per fragment
        A.u[q]     = *(const unsigned*)&zrow[kc + half * 16 + 2 * q];
        A.u[4 + q] = *(const unsigned*)&zrow[kc + half * 16 + 8 + 2 * q];
      }
      acc = __builtin_amdgcn_wmma_f32_16x16x32_bf16(false, A.v, false, Bf[t].v,
                                                    (short)0, acc, false, false);
    }
    #pragma unroll
    for (int v = 0; v < 8; ++v)                // D layout: VGPR v -> row v+8*half
      gbuf[s][r * 16 + v + 8 * half][ncol] = acc[v];
    __syncthreads();

    // ---- elementwise gates + state update (each thread owns 2 (n,j) pairs) ----
    unsigned short* hdst = hbuf + (size_t)(p ^ 1) * (N_*H_);
    #pragma unroll
    for (int rep = 0; rep < 2; ++rep) {
      int pidx = tid + rep * 256;
      int n = pidx >> 4, j = pidx & 15;
      float ig = fast_sigmoid(gbuf[0][n][j]);
      float fg = fast_sigmoid(gbuf[1][n][j]);
      float gg = fast_tanh(gbuf[2][n][j]);
      float og = fast_sigmoid(gbuf[3][n][j]);
      float c = fg * cbuf[n][j] + ig * gg;
      float h = og * fast_tanh(c);
      cbuf[n][j] = c;
      houts[n][j][l & 15] = h;
      hdst[n * H_ + hb0 + j] = f2bf(h);
    }

    // ---- every 16 steps flush 64B-contiguous rows of out (N,H,L) ----
    if ((l & 15) == 15) {
      __syncthreads();
      int l0 = l & ~15;
      #pragma unroll
      for (int rep = 0; rep < 2; ++rep) {
        int pidx = tid + rep * 256;
        int n = pidx >> 4, j = pidx & 15;
        float4* dst = (float4*)(out + (size_t)n * H_ * L_ + (size_t)(hb0 + j) * L_ + l0);
        const float4* src = (const float4*)&houts[n][j][0];
        dst[0] = src[0]; dst[1] = src[1]; dst[2] = src[2]; dst[3] = src[3];
      }
    }

    // ---- grid barrier across the 16 persistent blocks ----
    __syncthreads();
    if (tid == 0) {
      __threadfence();
      unsigned g = __atomic_load_n(&bar[1], __ATOMIC_RELAXED);
      unsigned arrived = __atomic_add_fetch(&bar[0], 1u, __ATOMIC_ACQ_REL);
      if (arrived == NBLK) {
        __atomic_store_n(&bar[0], 0u, __ATOMIC_RELAXED);
        __atomic_store_n(&bar[1], g + 1u, __ATOMIC_RELEASE);
      } else {
        while (__atomic_load_n(&bar[1], __ATOMIC_ACQUIRE) == g)
          __builtin_amdgcn_s_sleep(2);
      }
    }
    __syncthreads();
    p ^= 1;
  }
}

extern "C" void kernel_launch(void* const* d_in, const int* in_sizes, int n_in,
                              void* d_out, int out_size, void* d_ws, size_t ws_size,
                              hipStream_t stream) {
  const float* x   = (const float*)d_in[0];
  const float* Wih = (const float*)d_in[1];
  const float* Whh = (const float*)d_in[2];
  const float* bih = (const float*)d_in[3];
  const float* bhh = (const float*)d_in[4];
  float* out = (float*)d_out;

  char* ws = (char*)d_ws;
  unsigned short* xtb  = (unsigned short*)(ws);
  unsigned short* wb   = (unsigned short*)(ws + XTB_BYTES);
  float*          bias = (float*)(ws + XTB_BYTES + WB_BYTES);
  unsigned short* hbuf = (unsigned short*)(ws + XTB_BYTES + WB_BYTES + BIAS_BYTES);
  unsigned*       bar  = (unsigned*)(ws + BAR_OFF);

  prep_kernel<<<dim3((G_ * K_ + 255) / 256), dim3(256), 0, stream>>>(
      Wih, Whh, bih, bhh, wb, bias, hbuf, bar);
  transpose_kernel<<<dim3(L_ / 64, C_ / 64, N_), dim3(256), 0, stream>>>(x, xtb);
  lstm_kernel<<<dim3(NBLK), dim3(256), 0, stream>>>(xtb, wb, bias, hbuf, bar, out);
}
